// LiGRU_27771258536089
// MI455X (gfx1250) — matrix operand order repaired and checked by reading, not compiled, not run on hardware
//
#include <hip/hip_runtime.h>
#include <hip/hip_bf16.h>
#include <math.h>

// Problem constants (from the reference): B=32, T=512, D_IN=512, H=1024
#define B_   32
#define T_   512
#define DIN_ 512
#define H_   1024
#define G2H_ 2048          // 2*H (gates a|z)

typedef __attribute__((ext_vector_type(16))) __bf16 v16bf;
typedef __attribute__((ext_vector_type(8)))  float  v8f;
typedef __attribute__((ext_vector_type(4)))  unsigned int v4u;
typedef __attribute__((ext_vector_type(4)))  int v4i;
typedef __attribute__((ext_vector_type(8)))  int v8i;

union Frag16 { uint4 q[2]; v16bf v; };   // 16 bf16 = 32 bytes = 8 VGPRs

// LDS layout for the scan's U slice (written by TDM with padding):
//   stream of 512-ushort (1024 B) chunks, each followed by an 8-ushort (16 B) pad
//   -> element (row,k) at ushort index row*1040 + (k>>9)*520 + (k&511)
//   row stride 2080 B = 520 dwords == 8 banks/row shift; 16 B aligned chunks.
#define LROW 1040
#define LDS_BYTES ((size_t)64 * LROW * 2 + 64)

// ---------------------------------------------------------------- helpers
__device__ __forceinline__ unsigned short f2bf(float f) {
    unsigned u = __float_as_uint(f);
    unsigned r = u + 0x7FFFu + ((u >> 16) & 1u);   // round-to-nearest-even
    return (unsigned short)(r >> 16);
}

__device__ __forceinline__ void grid_sync(unsigned* cnt, unsigned* gen, unsigned nblk) {
    __builtin_amdgcn_fence(__ATOMIC_RELEASE, "agent");   // push h writes to device scope
    __syncthreads();
    if (threadIdx.x == 0) {
        unsigned g = __hip_atomic_load(gen, __ATOMIC_RELAXED, __HIP_MEMORY_SCOPE_AGENT);
        unsigned a = __hip_atomic_fetch_add(cnt, 1u, __ATOMIC_ACQ_REL, __HIP_MEMORY_SCOPE_AGENT);
        if (a == nblk - 1u) {
            __hip_atomic_store(cnt, 0u, __ATOMIC_RELAXED, __HIP_MEMORY_SCOPE_AGENT);
            __hip_atomic_store(gen, g + 1u, __ATOMIC_RELEASE, __HIP_MEMORY_SCOPE_AGENT);
        } else {
            while (__hip_atomic_load(gen, __ATOMIC_ACQUIRE, __HIP_MEMORY_SCOPE_AGENT) == g)
                __builtin_amdgcn_s_sleep(2);
        }
    }
    __syncthreads();
    __builtin_amdgcn_fence(__ATOMIC_ACQUIRE, "agent");   // invalidate near caches
}

// ---------------------------------------------------------------- fp32 -> bf16
__global__ void cvt_bf16(const float* __restrict__ s, unsigned short* __restrict__ d, int n4) {
    int i = blockIdx.x * blockDim.x + threadIdx.x;
    if (i < n4) {
        float4 f = ((const float4*)s)[i];
        ushort4 o;
        o.x = f2bf(f.x); o.y = f2bf(f.y); o.z = f2bf(f.z); o.w = f2bf(f.w);
        ((ushort4*)d)[i] = o;
    }
}

// ---------------------------------------------------------------- state init
__global__ void init_state(float* hf, unsigned short* hb, unsigned* bar) {
    int i = blockIdx.x * blockDim.x + threadIdx.x;
    if (i < B_ * H_) { hf[i] = 0.0f; hb[i] = 0; }
    if (i < 2) bar[i] = 0;
}

// ---------------------------------------------------------------- input projection
// C[M, 2048] = A[M,K](bf16 rm) * W[2048,K](bf16 rm)^T   (both K-contiguous)
// One wave computes a 16x64 strip: A-frag reused across 4 B-frags.
__global__ void __launch_bounds__(256)
proj_wmma(const unsigned short* __restrict__ A,
          const unsigned short* __restrict__ W,
          float* __restrict__ C, int M, int K) {
    const int N    = G2H_;
    const int wave = blockIdx.x * (blockDim.x >> 5) + (threadIdx.x >> 5);
    const int lane = threadIdx.x & 31;
    const int lm   = lane & 15;
    const int hi   = lane >> 4;
    const int mtiles = M >> 4;
    const int mt = wave % mtiles;
    const int nt = wave / mtiles;          // 0..31 -> 64-wide column strip
    const int m0 = mt << 4, n0 = nt << 6;

    const unsigned short* arow = A + (size_t)(m0 + lm) * K + hi * 8;

    v8f z8 = {0.f,0.f,0.f,0.f,0.f,0.f,0.f,0.f};
    v8f acc[4] = { z8, z8, z8, z8 };

    for (int kb = 0; kb < K; kb += 32) {
        __builtin_prefetch(arow + kb + 128, 0, 1);      // global_prefetch_b8
        Frag16 a;
        a.q[0] = *(const uint4*)(arow + kb);            // K = kb + hi*8 + 0..7
        a.q[1] = *(const uint4*)(arow + kb + 16);       // K = kb + 16 + hi*8 + 0..7
#pragma unroll
        for (int t = 0; t < 4; ++t) {
            const unsigned short* brow = W + (size_t)(n0 + t * 16 + lm) * K + kb + hi * 8;
            Frag16 b;
            b.q[0] = *(const uint4*)(brow);
            b.q[1] = *(const uint4*)(brow + 16);
            acc[t] = __builtin_amdgcn_wmma_f32_16x16x32_bf16(
                         false, a.v, false, b.v, (short)0, acc[t], false, false);
        }
    }
#pragma unroll
    for (int t = 0; t < 4; ++t)
#pragma unroll
        for (int r = 0; r < 8; ++r)
            C[(size_t)(m0 + r + hi * 8) * N + (n0 + t * 16 + lm)] = acc[t][r];
}

// ---------------------------------------------------------------- recurrent scan
// Persistent: 32 WGs x 128 threads (4 waves). WG `wg` owns hidden columns
// [wg*32, wg*32+32): U rows wg*32..+32 ("a") and H+wg*32..+32 ("z") staged into
// LDS by the Tensor Data Mover. Wave w: M-tile = w&1, N-subtile = w>>1.
__global__ void __launch_bounds__(128)
ligru_scan(const float* __restrict__ wbuf,          // [B,T,2H] f32 gate pre-proj
           const unsigned short* __restrict__ U,    // [2H,H] bf16
           float* hfA, float* hfB,                  // [B,H] f32 double buffer
           unsigned short* hbA, unsigned short* hbB,// [B,H] bf16 double buffer
           unsigned short* out_bf,                  // layer0: [B,T,H] bf16 (else null)
           float* out_f,                            // layer1: [B,T,H] f32  (else null)
           float* __restrict__ last_dst,            // [B,H] f32 (final h)
           unsigned* bar) {
    extern __shared__ unsigned short lds[];

    const int wg   = blockIdx.x;                    // 0..31
    const int tid  = threadIdx.x;
    const int lane = tid & 31;
    const int wv   = tid >> 5;
    const int lm   = lane & 15;
    const int hi   = lane >> 4;
    const int m0   = (wv & 1) << 4;
    const int nt   = wv >> 1;

#if __has_builtin(__builtin_amdgcn_tensor_load_to_lds)
    // ---- TDM staging: two 2-D tiles (1024 x 32 bf16), DMA'd with LDS padding.
    if (wv == 0) {
        unsigned ldsb = (unsigned)(size_t)(&lds[0]);   // LDS aperture: addr[31:0] = byte offset
        unsigned long long ga = (unsigned long long)(const void*)(U + (size_t)(wg * 32) * H_);
        unsigned long long gz = (unsigned long long)(const void*)(U + (size_t)(H_ + wg * 32) * H_);
        // D# group0: count=1 | lds_addr | global_addr | type=2
        v4u g0a = { 1u, ldsb,
                    (unsigned)ga, (unsigned)(ga >> 32) | (2u << 30) };
        v4u g0z = { 1u, ldsb + (unsigned)(32 * LROW * 2),
                    (unsigned)gz, (unsigned)(gz >> 32) | (2u << 30) };
        // D# group1: data_size=2B, pad_enable, pad_interval=7 (256 dw), pad_amount=1 (2 dw),
        //            tensor_dim0=1024, tensor_dim1=32, tile_dim0=1024, tile_dim1=32,
        //            tensor_dim0_stride=1024
        v8i g1 = { (int)((1u << 16) | (1u << 20) | (7u << 22) | (1u << 25)),
                   (int)(1024u << 16),    // tensor_dim0[15:0] @ bits 63:48
                   (int)(32u   << 16),    // tensor_dim0 hi=0 | tensor_dim1[15:0]
                   (int)(1024u << 16),    // tensor_dim1 hi=0 | tile_dim0
                   32,                    // tile_dim1 | tile_dim2=0
                   1024,                  // tensor_dim0_stride[31:0]
                   0, 0 };                // stride hi / tensor_dim1_stride = 0
        v4i z4 = {0, 0, 0, 0};
#if defined(__clang_major__) && (__clang_major__ >= 23)
        v8i z8i = {0, 0, 0, 0, 0, 0, 0, 0};
        __builtin_amdgcn_tensor_load_to_lds(g0a, g1, z4, z4, z8i, 0);
        __builtin_amdgcn_tensor_load_to_lds(g0z, g1, z4, z4, z8i, 0);
#else
        __builtin_amdgcn_tensor_load_to_lds(g0a, g1, z4, z4, 0);
        __builtin_amdgcn_tensor_load_to_lds(g0z, g1, z4, z4, 0);
#endif
        __builtin_amdgcn_s_wait_tensorcnt(0);
    }
    __syncthreads();
#else
    // ---- fallback: manual staging with the same padded layout
    for (int idx = tid; idx < 64 * 128; idx += 128) {
        int row = idx >> 7;
        int c8  = (idx & 127) << 3;
        int grow = (row < 32) ? (wg * 32 + row) : (H_ + wg * 32 + (row - 32));
        *(uint4*)(&lds[row * LROW + ((c8 >> 9) * 520) + (c8 & 511)]) =
            *(const uint4*)(U + (size_t)grow * H_ + c8);
    }
    __syncthreads();
#endif

    float*          hf_cur = hfA; float*          hf_nxt = hfB;
    unsigned short* hb_cur = hbA; unsigned short* hb_nxt = hbB;

    const int jl  = nt * 16 + lm;           // local column 0..31
    const int col = wg * 32 + jl;           // global hidden column
    const unsigned short* lba = lds + jl * LROW;          // B-frag row for "a"
    const unsigned short* lbz = lds + (32 + jl) * LROW;   // B-frag row for "z"

    v8f z8 = {0.f,0.f,0.f,0.f,0.f,0.f,0.f,0.f};

    for (int t = 0; t < T_; ++t) {
        v8f aacc = z8, zacc = z8;
        const unsigned short* hrow = hb_cur + (m0 + lm) * H_ + hi * 8;
#pragma unroll 4
        for (int kb = 0; kb < H_; kb += 32) {
            const int lk = ((kb >> 9) * 520) + (kb & 511) + hi * 8;  // padded LDS offset
            Frag16 a;
            a.q[0] = *(const uint4*)(hrow + kb);
            a.q[1] = *(const uint4*)(hrow + kb + 16);
            Frag16 ba, bz;
            ba.q[0] = *(const uint4*)(lba + lk); ba.q[1] = *(const uint4*)(lba + lk + 16);
            bz.q[0] = *(const uint4*)(lbz + lk); bz.q[1] = *(const uint4*)(lbz + lk + 16);
            aacc = __builtin_amdgcn_wmma_f32_16x16x32_bf16(
                       false, a.v, false, ba.v, (short)0, aacc, false, false);
            zacc = __builtin_amdgcn_wmma_f32_16x16x32_bf16(
                       false, a.v, false, bz.v, (short)0, zacc, false, false);
        }
#pragma unroll
        for (int r = 0; r < 8; ++r) {
            const int b = m0 + r + hi * 8;                    // batch row
            const size_t wrow = ((size_t)b * T_ + t) * G2H_;
            float apre = wbuf[wrow + col]       + aacc[r];
            float zpre = wbuf[wrow + H_ + col]  + zacc[r];
            float zs   = 1.0f / (1.0f + __expf(-zpre));
            float hc   = tanhf(apre);
            float hold = hf_cur[b * H_ + col];
            float hn   = zs * hold + (1.0f - zs) * hc;
            hf_nxt[b * H_ + col] = hn;
            hb_nxt[b * H_ + col] = f2bf(hn);
            const size_t orow = ((size_t)b * T_ + t) * H_ + col;
            if (out_bf) out_bf[orow] = f2bf(hn);
            else        out_f[orow]  = hn;
            if (t == T_ - 1) last_dst[b * H_ + col] = hn;
        }
        grid_sync(bar, bar + 1, gridDim.x);
        float* tf = hf_cur; hf_cur = hf_nxt; hf_nxt = tf;
        unsigned short* tb = hb_cur; hb_cur = hb_nxt; hb_nxt = tb;
    }
}

// ---------------------------------------------------------------- launch
extern "C" void kernel_launch(void* const* d_in, const int* in_sizes, int n_in,
                              void* d_out, int out_size, void* d_ws, size_t ws_size,
                              hipStream_t stream) {
    (void)in_sizes; (void)n_in; (void)out_size; (void)ws_size;
    const float* x  = (const float*)d_in[0];
    const float* W0 = (const float*)d_in[1];
    const float* U0 = (const float*)d_in[2];
    const float* W1 = (const float*)d_in[3];
    const float* U1 = (const float*)d_in[4];
    float* out = (float*)d_out;

    // Workspace carve-out (~200 MB total)
    char* ws = (char*)d_ws;
    size_t off = 0;
    auto carve = [&](size_t bytes) -> char* {
        char* p = ws + off;
        off = (off + bytes + 255) & ~(size_t)255;
        return p;
    };
    unsigned short* xbf  = (unsigned short*)carve((size_t)B_ * T_ * DIN_ * 2);
    unsigned short* w0bf = (unsigned short*)carve((size_t)G2H_ * DIN_ * 2);
    unsigned short* u0bf = (unsigned short*)carve((size_t)G2H_ * H_ * 2);
    unsigned short* w1bf = (unsigned short*)carve((size_t)G2H_ * H_ * 2);
    unsigned short* u1bf = (unsigned short*)carve((size_t)G2H_ * H_ * 2);
    float*          wbuf = (float*)carve((size_t)B_ * T_ * G2H_ * 4);   // reused per layer
    unsigned short* out0 = (unsigned short*)carve((size_t)B_ * T_ * H_ * 2);
    float*          hfA  = (float*)carve((size_t)B_ * H_ * 4);
    float*          hfB  = (float*)carve((size_t)B_ * H_ * 4);
    unsigned short* hbA  = (unsigned short*)carve((size_t)B_ * H_ * 2);
    unsigned short* hbB  = (unsigned short*)carve((size_t)B_ * H_ * 2);
    unsigned*       bar  = (unsigned*)carve(256);

    (void)hipFuncSetAttribute(reinterpret_cast<const void*>(ligru_scan),
                              hipFuncAttributeMaxDynamicSharedMemorySize,
                              (int)LDS_BYTES);

    auto cvt = [&](const float* s, unsigned short* d, size_t n) {
        int n4 = (int)(n / 4);
        cvt_bf16<<<(n4 + 255) / 256, 256, 0, stream>>>(s, d, n4);
    };
    cvt(x,  xbf,  (size_t)B_ * T_ * DIN_);
    cvt(W0, w0bf, (size_t)G2H_ * DIN_);
    cvt(U0, u0bf, (size_t)G2H_ * H_);
    cvt(W1, w1bf, (size_t)G2H_ * H_);
    cvt(U1, u1bf, (size_t)G2H_ * H_);

    // waves = (M/16)*(2048/64) = 1024*32; 8 waves/block -> 4096 blocks
    const int projBlocks = ((B_ * T_ / 16) * (G2H_ / 64)) / 8;

    // ---- layer 0
    proj_wmma<<<projBlocks, 256, 0, stream>>>(xbf, w0bf, wbuf, B_ * T_, DIN_);
    init_state<<<(B_ * H_ + 255) / 256, 256, 0, stream>>>(hfA, hbA, bar);
    ligru_scan<<<32, 128, LDS_BYTES, stream>>>(wbuf, u0bf, hfA, hfB, hbA, hbB,
                                               out0, nullptr,
                                               out + (size_t)B_ * T_ * H_, bar);
    // ---- layer 1
    proj_wmma<<<projBlocks, 256, 0, stream>>>(out0, w1bf, wbuf, B_ * T_, H_);
    init_state<<<(B_ * H_ + 255) / 256, 256, 0, stream>>>(hfA, hbA, bar);
    ligru_scan<<<32, 128, LDS_BYTES, stream>>>(wbuf, u1bf, hfA, hfB, hbA, hbB,
                                               nullptr, out,
                                               out + (size_t)B_ * T_ * H_ + (size_t)B_ * H_, bar);
}
